// EncoderBlock_75900662054951
// MI455X (gfx1250) — matrix operand order repaired and checked by reading, not compiled, not run on hardware
//
#include <hip/hip_runtime.h>

// ---------------------------------------------------------------------------
// Transformer encoder block for MI455X (gfx1250, wave32, WMMA).
// All GEMMs go through one WMMA bf16 kernel: C[M,N] = A[M,K] * Bt[N,K]^T.
//  - fp32 global sources: staged to LDS with in-flight RNE->bf16 conversion.
//  - bf16 global sources: staged with the CDNA5 async copy engine
//    (global_load_async_to_lds_b128, ASYNCcnt), no VGPR bounce.
// ---------------------------------------------------------------------------

constexpr int Bc  = 2;
constexpr int Sc  = 4096;
constexpr int Dc  = 512;
constexpr int Hc  = 8;
constexpr int FFc = 2048;
constexpr int HDc = 64;

typedef __attribute__((ext_vector_type(8)))  float   v8f;
typedef __attribute__((ext_vector_type(8)))  __bf16  v8bf;
typedef __attribute__((ext_vector_type(16))) __bf16  v16bf;

__device__ __forceinline__ unsigned short f2bf(float f) {
    unsigned int u = __float_as_uint(f);
    u += 0x7FFFu + ((u >> 16) & 1u);           // round-to-nearest-even
    return (unsigned short)(u >> 16);
}

// fp32 global -> bf16 LDS staging, 4 elements per call
__device__ __forceinline__ void ld_cvt4(unsigned short* dst, const float* src) {
    float4 t = *(const float4*)src;
    ushort4 o;
    o.x = f2bf(t.x); o.y = f2bf(t.y); o.z = f2bf(t.z); o.w = f2bf(t.w);
    *(ushort4*)dst = o;
}

// bf16 global -> LDS, 16 bytes per lane via the async copy engine.
// VDST operand = LDS byte offset = low 32 bits of the generic pointer
// (ISA 10.2).
__device__ __forceinline__ void async_cp16(unsigned short* ldsdst, const unsigned short* src) {
    unsigned loff = (unsigned)(size_t)ldsdst;
    unsigned long long gaddr = (unsigned long long)(size_t)src;
    asm volatile("global_load_async_to_lds_b128 %0, %1, off"
                 :: "v"(loff), "v"(gaddr) : "memory");
}
__device__ __forceinline__ void async_wait0() {
    asm volatile("s_wait_asynccnt 0x0" ::: "memory");
}

// build a 16-half WMMA fragment from two 16B LDS chunks
__device__ __forceinline__ v16bf frag2(const unsigned short* p0, const unsigned short* p1) {
    v8bf lo = *(const v8bf*)p0;
    v8bf hi = *(const v8bf*)p1;
    return __builtin_shufflevector(lo, hi, 0,1,2,3,4,5,6,7,8,9,10,11,12,13,14,15);
}

enum { EPI_F32 = 0, EPI_BF16_RELU = 1, EPI_QK = 2, EPI_SCORES = 3, EPI_CTX = 4 };

// ---------------------------------------------------------------------------
// Generic WMMA GEMM: C = A[M,K] * Bt[N,K]^T  (A,Bt row-major, K contiguous).
// 256 threads = 8 wave32s. BK = 32 (one v_wmma_f32_16x16x32_bf16 per k-step).
// SA/SB select fp32 (convert on load) or bf16 (async copy) global sources.
// ---------------------------------------------------------------------------
template<typename SA, typename SB, int BM, int BN, int WAVES_M, int WAVES_N, int EPI>
__global__ __launch_bounds__(256)
void enc_gemm_wmma(const void* __restrict__ Ag, const void* __restrict__ Bg,
                   void* __restrict__ Og, void* __restrict__ Og2,
                   int M, int N, int K,
                   long long strideAz, long long strideBz, float alpha)
{
    constexpr int BK   = 32;
    constexpr int LDST = BK + 8;               // 80B row stride: 16B-aligned frags, bank-spread
    constexpr int WM   = BM / WAVES_M;
    constexpr int WN   = BN / WAVES_N;
    constexpr int MT   = WM / 16;
    constexpr int NT   = WN / 16;
    constexpr bool A_BF = (sizeof(SA) == 2);
    constexpr bool B_BF = (sizeof(SB) == 2);

    __shared__ unsigned short As[BM * LDST];
    __shared__ unsigned short Bs[BN * LDST];

    const int tid  = threadIdx.x;
    const int lane = tid & 31;
    const int wave = tid >> 5;
    const int wm   = wave % WAVES_M;
    const int wn   = wave / WAVES_M;
    const int z    = blockIdx.z;

    const SA* A  = (const SA*)Ag + (size_t)z * (size_t)strideAz;
    const SB* Bm = (const SB*)Bg + (size_t)z * (size_t)strideBz;

    const int rowBase = blockIdx.y * BM;
    const int colBase = blockIdx.x * BN;

    const v8f zacc = {0.f,0.f,0.f,0.f,0.f,0.f,0.f,0.f};
    v8f acc[MT][NT];
#pragma unroll
    for (int i = 0; i < MT; ++i)
#pragma unroll
        for (int j = 0; j < NT; ++j) acc[i][j] = zacc;

    const int fm  = lane >> 4;           // half-wave select
    const int fnl = lane & 15;
    const int koA = fm * 8;              // A chunk K-offsets: koA and 16+koA
    const int koB = fm * 8;              // B chunk K-offsets: koB and 16+koB

    for (int k0 = 0; k0 < K; k0 += BK) {
        if (k0 + BK < K) {               // global_prefetch_b8 of next K-slab
            __builtin_prefetch(A  + (size_t)(rowBase + (tid % BM)) * K + k0 + BK, 0, 1);
            __builtin_prefetch(Bm + (size_t)(colBase + (tid % BN)) * K + k0 + BK, 0, 1);
        }
        // ---- stage A tile ----
        if constexpr (A_BF) {
#pragma unroll
            for (int j = 0; j < (BM * BK) / (256 * 8); ++j) {
                int flat = (tid + j * 256) * 8;
                int r = flat / BK, c = flat % BK;
                async_cp16(&As[r * LDST + c],
                           (const unsigned short*)A + (size_t)(rowBase + r) * K + (k0 + c));
            }
        } else {
#pragma unroll
            for (int j = 0; j < (BM * BK) / (256 * 4); ++j) {
                int flat = (tid + j * 256) * 4;
                int r = flat / BK, c = flat % BK;
                ld_cvt4(&As[r * LDST + c],
                        (const float*)A + (size_t)(rowBase + r) * K + (k0 + c));
            }
        }
        // ---- stage B tile ----
        if constexpr (B_BF) {
#pragma unroll
            for (int j = 0; j < (BN * BK) / (256 * 8); ++j) {
                int flat = (tid + j * 256) * 8;
                int r = flat / BK, c = flat % BK;
                async_cp16(&Bs[r * LDST + c],
                           (const unsigned short*)Bm + (size_t)(colBase + r) * K + (k0 + c));
            }
        } else {
#pragma unroll
            for (int j = 0; j < (BN * BK) / (256 * 4); ++j) {
                int flat = (tid + j * 256) * 4;
                int r = flat / BK, c = flat % BK;
                ld_cvt4(&Bs[r * LDST + c],
                        (const float*)Bm + (size_t)(colBase + r) * K + (k0 + c));
            }
        }
        if constexpr (A_BF || B_BF) async_wait0();   // drain ASYNCcnt before the barrier
        __syncthreads();

        v16bf af[MT];
#pragma unroll
        for (int i = 0; i < MT; ++i) {
            const unsigned short* rp = &As[(wm * WM + i * 16 + fnl) * LDST];
            af[i] = frag2(rp + koA, rp + 16 + koA);
        }
        v16bf bfr[NT];
#pragma unroll
        for (int j = 0; j < NT; ++j) {
            const unsigned short* rp = &Bs[(wn * WN + j * 16 + fnl) * LDST];
            bfr[j] = frag2(rp + koB, rp + 16 + koB);
        }
#pragma unroll
        for (int i = 0; i < MT; ++i)
#pragma unroll
            for (int j = 0; j < NT; ++j)
                acc[i][j] = __builtin_amdgcn_wmma_f32_16x16x32_bf16(
                    false, af[i], false, bfr[j], (short)0, acc[i][j], false, false);
        __syncthreads();
    }

    // ---- epilogue: C/D layout is m = r + (lane>>4)*8, n = lane&15 ----------
#pragma unroll
    for (int i = 0; i < MT; ++i)
#pragma unroll
        for (int j = 0; j < NT; ++j)
#pragma unroll
            for (int r = 0; r < 8; ++r) {
                int gm = rowBase + wm * WM + i * 16 + fm * 8 + r;
                int gn = colBase + wn * WN + j * 16 + fnl;
                float v = acc[i][j][r];
                if constexpr (EPI == EPI_F32) {
                    ((float*)Og)[(size_t)gm * N + gn] = v * alpha;
                } else if constexpr (EPI == EPI_BF16_RELU) {
                    ((unsigned short*)Og)[(size_t)gm * N + gn] = f2bf(fmaxf(v, 0.f));
                } else if constexpr (EPI == EPI_QK) {
                    // y[b,s,h*HD+hd] -> q[b,h,s,hd] (and optionally v^T[b,h,hd,s])
                    int b = gm / Sc, s = gm % Sc;
                    int h = gn / HDc, hd = gn % HDc;
                    unsigned short bv = f2bf(v);
                    ((unsigned short*)Og)[(((size_t)b * Hc + h) * Sc + s) * HDc + hd] = bv;
                    if (Og2)
                        ((unsigned short*)Og2)[(((size_t)b * Hc + h) * HDc + hd) * Sc + s] = bv;
                } else if constexpr (EPI == EPI_SCORES) {
                    ((float*)Og)[(size_t)z * Sc * Sc + (size_t)gm * Sc + gn] = v * alpha;
                } else { // EPI_CTX: per-head [S,HD] -> [B,S,D]
                    int b = z / Hc, h = z % Hc;
                    ((unsigned short*)Og)[((size_t)(b * Sc + gm)) * Dc + h * HDc + gn] = f2bf(v);
                }
            }
}

// ---------------------------------------------------------------------------
// Row softmax over 4096 elements, one block (8 waves) per row, in place.
// ---------------------------------------------------------------------------
__global__ __launch_bounds__(256)
void enc_softmax_rows(float* __restrict__ attn)
{
    const size_t row = blockIdx.x;
    float* p = attn + row * (size_t)Sc;
    const int tid = threadIdx.x;
    __shared__ float red[256];

    float v[16];
    float mx = -3.4e38f;
#pragma unroll
    for (int i = 0; i < 16; ++i) { v[i] = p[tid + i * 256]; mx = fmaxf(mx, v[i]); }
    red[tid] = mx; __syncthreads();
    for (int s = 128; s > 0; s >>= 1) {
        if (tid < s) red[tid] = fmaxf(red[tid], red[tid + s]);
        __syncthreads();
    }
    mx = red[0]; __syncthreads();

    float sum = 0.f;
#pragma unroll
    for (int i = 0; i < 16; ++i) { v[i] = __expf(v[i] - mx); sum += v[i]; }
    red[tid] = sum; __syncthreads();
    for (int s = 128; s > 0; s >>= 1) {
        if (tid < s) red[tid] += red[tid + s];
        __syncthreads();
    }
    const float inv = 1.0f / red[0];
#pragma unroll
    for (int i = 0; i < 16; ++i) p[tid + i * 256] = v[i] * inv;
}

// ---------------------------------------------------------------------------
// out = LayerNorm(xa + xb) * g + b, one block per row of 512.
// ---------------------------------------------------------------------------
__global__ __launch_bounds__(256)
void enc_add_ln(const float* __restrict__ xa, const float* __restrict__ xb,
                const float* __restrict__ g, const float* __restrict__ bb,
                float* __restrict__ out)
{
    const size_t base = (size_t)blockIdx.x * Dc;
    const int tid = threadIdx.x;
    __shared__ float red[256];

    const int i0 = tid, i1 = tid + 256;
    float s0 = xa[base + i0] + xb[base + i0];
    float s1 = xa[base + i1] + xb[base + i1];

    red[tid] = s0 + s1; __syncthreads();
    for (int s = 128; s > 0; s >>= 1) {
        if (tid < s) red[tid] += red[tid + s];
        __syncthreads();
    }
    const float mu = red[0] * (1.0f / Dc); __syncthreads();

    red[tid] = s0 * s0 + s1 * s1; __syncthreads();
    for (int s = 128; s > 0; s >>= 1) {
        if (tid < s) red[tid] += red[tid + s];
        __syncthreads();
    }
    const float var = red[0] * (1.0f / Dc) - mu * mu;
    const float rs  = rsqrtf(var + 1e-5f);

    out[base + i0] = (s0 - mu) * rs * g[i0] + bb[i0];
    out[base + i1] = (s1 - mu) * rs * g[i1] + bb[i1];
}

// ---------------------------------------------------------------------------
extern "C" void kernel_launch(void* const* d_in, const int* in_sizes, int n_in,
                              void* d_out, int out_size, void* d_ws, size_t ws_size,
                              hipStream_t stream)
{
    (void)in_sizes; (void)n_in; (void)out_size; (void)ws_size;

    const float* x    = (const float*)d_in[0];
    const float* Wq   = (const float*)d_in[2];
    const float* Wk   = (const float*)d_in[3];
    const float* Wfc  = (const float*)d_in[5];
    const float* Wff1 = (const float*)d_in[6];
    const float* Wff2 = (const float*)d_in[7];
    const float* g1   = (const float*)d_in[8];
    const float* b1   = (const float*)d_in[9];
    const float* g2   = (const float*)d_in[10];
    const float* b2   = (const float*)d_in[11];

    float* out_feed = (float*)d_out;
    float* attn     = out_feed + (size_t)Bc * Sc * Dc;   // [B,H,S,S] region of d_out

    // workspace arena (regions reused once their producers/consumers are done)
    char* ws = (char*)d_ws;
    unsigned short* q_bf   = (unsigned short*)(ws + 0);          //  8 MiB bf16 [B,H,S,HD]
    unsigned short* k_bf   = (unsigned short*)(ws + 8388608);    //  8 MiB bf16 [B,H,S,HD]
    unsigned short* kT_bf  = (unsigned short*)(ws + 16777216);   //  8 MiB bf16 [B,H,HD,S]  (v == k)
    unsigned short* ctx_bf = (unsigned short*)(ws + 25165824);   //  8 MiB bf16 [B,S,D]
    float*          fc_f   = (float*)(ws + 0);                   // 16 MiB (reuses q+k)
    float*          attres = (float*)(ws + 33554432);            // 16 MiB fp32 [B,S,D]
    unsigned short* ff1_bf = (unsigned short*)(ws + 50331648);   // 32 MiB bf16 [B,S,FF]
    float*          ff2_f  = (float*)(ws + 0);                   // 16 MiB (reuses fc)

    const dim3 blk(256);
    const int M = Bc * Sc;                                       // 8192
    const float inv_sqrt_hd = 0.125f;                            // 1/sqrt(D/H)

    // Q = x @ Wq^T  -> q_bf [B,H,S,HD]
    enc_gemm_wmma<float, float, 128, 128, 2, 4, EPI_QK>
        <<<dim3(Dc / 128, M / 128, 1), blk, 0, stream>>>(
            x, Wq, q_bf, nullptr, M, Dc, Dc, 0, 0, 1.f);
    // K = x @ Wk^T  -> k_bf and kT_bf (V uses W_K per reference bug)
    enc_gemm_wmma<float, float, 128, 128, 2, 4, EPI_QK>
        <<<dim3(Dc / 128, M / 128, 1), blk, 0, stream>>>(
            x, Wk, k_bf, kT_bf, M, Dc, Dc, 0, 0, 1.f);
    // scores = q @ k^T / 8 -> attn region (fp32, raw); bf16 operands -> async staging
    enc_gemm_wmma<unsigned short, unsigned short, 128, 128, 2, 4, EPI_SCORES>
        <<<dim3(Sc / 128, Sc / 128, Bc * Hc), blk, 0, stream>>>(
            q_bf, k_bf, attn, nullptr, Sc, Sc, HDc,
            (long long)Sc * HDc, (long long)Sc * HDc, inv_sqrt_hd);
    // softmax rows in place
    enc_softmax_rows<<<Bc * Hc * Sc, blk, 0, stream>>>(attn);
    // context = attn @ v  (v^T stored) -> ctx_bf [B,S,D]
    enc_gemm_wmma<float, unsigned short, 128, 64, 4, 2, EPI_CTX>
        <<<dim3(1, Sc / 128, Bc * Hc), blk, 0, stream>>>(
            attn, kT_bf, ctx_bf, nullptr, Sc, HDc, Sc,
            (long long)Sc * Sc, (long long)HDc * Sc, 1.f);
    // fc = context @ Wfc^T -> fp32
    enc_gemm_wmma<unsigned short, float, 128, 128, 2, 4, EPI_F32>
        <<<dim3(Dc / 128, M / 128, 1), blk, 0, stream>>>(
            ctx_bf, Wfc, fc_f, nullptr, M, Dc, Dc, 0, 0, 1.f);
    // att_result = LN(x + fc)
    enc_add_ln<<<M, blk, 0, stream>>>(x, fc_f, g1, b1, attres);
    // ff1 = relu(att_result @ Wff1^T) -> bf16
    enc_gemm_wmma<float, float, 128, 128, 2, 4, EPI_BF16_RELU>
        <<<dim3(FFc / 128, M / 128, 1), blk, 0, stream>>>(
            attres, Wff1, ff1_bf, nullptr, M, FFc, Dc, 0, 0, 1.f);
    // ff2 = ff1 @ Wff2^T -> fp32
    enc_gemm_wmma<unsigned short, float, 128, 128, 2, 4, EPI_F32>
        <<<dim3(Dc / 128, M / 128, 1), blk, 0, stream>>>(
            ff1_bf, Wff2, ff2_f, nullptr, M, Dc, FFc, 0, 0, 1.f);
    // feed_result = LN(att_result + ff2) -> d_out
    enc_add_ln<<<M, blk, 0, stream>>>(attres, ff2_f, g2, b2, out_feed);
}
